// YoloMaskNet_28260884807838
// MI455X (gfx1250) — matrix operand order repaired and checked by reading, not compiled, not run on hardware
//
#include <hip/hip_runtime.h>
#include <hip/hip_fp16.h>

typedef __attribute__((ext_vector_type(16))) _Float16 v16h;
typedef __attribute__((ext_vector_type(8)))  _Float16 v8h;
typedef __attribute__((ext_vector_type(8)))  float    v8f;
typedef __attribute__((ext_vector_type(4)))  float    v4f;

// ---------------------------------------------------------------------------
// Implicit-GEMM convolution with WMMA f16 (NHWC fp32 in, weights pre-converted
// to f16 [Cout][k*k*Cin]). Fused BN-scale+bias+leakyReLU (or bias only) and
// optional residual add. Two variants:
//   - fast:    Cin % 32 == 0 -> affine (kh,kw,cin-block) loop, b128 staging
//   - generic: any Cin (used only for the 3-channel stem conv)
// Block: 128 threads = 4 waves; block tile 64x64; each wave 32x32 via
// 2x2 v_wmma_f32_16x16x32_f16 fragments.
// ---------------------------------------------------------------------------
struct ConvParams {
  const float*    x;         // (B,H,W,Cin) fp32
  const _Float16* wT;        // [Cout][Ktot] f16
  const float*    scale;     // [Cout] or null
  const float*    bias;      // [Cout]
  const float*    residual;  // (M,Cout) or null
  float*          y;         // (M,Cout)
  int B, H, W, Cin, Cout, Ksz, stride, pad, Ho, Wo, Ktot, M, bnAct;
};

#define LDSF 48  // fast-path LDS row stride in halves (96B: keeps v16h aligned)

__device__ __forceinline__ void conv_epilogue(const ConvParams& p, v8f acc[2][2],
                                              int tileM, int tileN, int warpM,
                                              int warpN, int lane) {
  const int lcol   = lane & 15;
  const int rowOff = (lane >> 4) * 8;
  #pragma unroll
  for (int i = 0; i < 2; ++i) {
    #pragma unroll
    for (int j = 0; j < 2; ++j) {
      int nIdx = tileN + warpN * 32 + j * 16 + lcol;
      #pragma unroll
      for (int rr = 0; rr < 8; ++rr) {
        int mIdx = tileM + warpM * 32 + i * 16 + rr + rowOff;
        if (mIdx < p.M && nIdx < p.Cout) {
          float v = acc[i][j][rr];
          if (p.bnAct) {
            v = v * p.scale[nIdx] + p.bias[nIdx];
            v = (v > 0.0f) ? v : 0.1f * v;
          } else {
            v = v + p.bias[nIdx];
          }
          if (p.residual) v += p.residual[(size_t)mIdx * p.Cout + nIdx];
          p.y[(size_t)mIdx * p.Cout + nIdx] = v;
        }
      }
    }
  }
}

// ---------------- fast path: Cin % 32 == 0 ----------------
__global__ __launch_bounds__(128)
void yolo_conv_wmma_fast(ConvParams p) {
  __shared__ __align__(64) _Float16 As[64 * LDSF];
  __shared__ __align__(64) _Float16 Bs[64 * LDSF];

  const int tid   = threadIdx.x;
  const int lane  = tid & 31;
  const int wave  = tid >> 5;
  const int warpM = wave >> 1;
  const int warpN = wave & 1;
  const int tileM = blockIdx.x * 64;
  const int tileN = blockIdx.y * 64;

  const int r  = tid >> 1;          // staged row 0..63
  const int c0 = (tid & 1) * 16;    // k sub-range 0 or 16

  const int m = tileM + r;
  const bool mvalid = (m < p.M);
  int ow = 0, oh = 0, bb = 0;
  if (mvalid) {
    ow = m % p.Wo;
    int t2 = m / p.Wo;
    oh = t2 % p.Ho;
    bb = t2 / p.Ho;
  }
  const int ihs = oh * p.stride - p.pad;
  const int iws = ow * p.stride - p.pad;
  const int n   = tileN + r;
  const bool nvalid = (n < p.Cout);
  const _Float16* wRow = p.wT + (size_t)n * p.Ktot;

  v8f zero = {0.f,0.f,0.f,0.f,0.f,0.f,0.f,0.f};
  v8f acc[2][2];
  acc[0][0] = zero; acc[0][1] = zero; acc[1][0] = zero; acc[1][1] = zero;

  const int lrow = lane & 15;
  const int sel  = lane >> 4;

  for (int kh = 0; kh < p.Ksz; ++kh) {
    const int ih = ihs + kh;
    const bool ihok = (ih >= 0) & (ih < p.H);
    for (int kw = 0; kw < p.Ksz; ++kw) {
      const int iw = iws + kw;
      const bool ok = mvalid & ihok & (iw >= 0) & (iw < p.W);
      const float* srcRow =
          p.x + (size_t)((bb * p.H + (ok ? ih : 0)) * p.W + (ok ? iw : 0)) * p.Cin;
      const int kbase = (kh * p.Ksz + kw) * p.Cin;

      for (int cb = 0; cb < p.Cin; cb += 32) {
        // ---- stage A: 16 contiguous fp32 (64B aligned) -> f16 LDS b128 ----
        {
          v4f z4 = {0.f,0.f,0.f,0.f};
          v4f f0 = z4, f1 = z4, f2 = z4, f3 = z4;
          if (ok) {
            const v4f* s4 = (const v4f*)(srcRow + cb + c0);
            f0 = s4[0]; f1 = s4[1]; f2 = s4[2]; f3 = s4[3];
          }
          v16h tmp;
          #pragma unroll
          for (int j = 0; j < 4; ++j) {
            tmp[j]      = (_Float16)f0[j];
            tmp[4 + j]  = (_Float16)f1[j];
            tmp[8 + j]  = (_Float16)f2[j];
            tmp[12 + j] = (_Float16)f3[j];
          }
          *(v16h*)(&As[r * LDSF + c0]) = tmp;
        }
        // ---- stage B: 16 contiguous f16 (32B aligned) ----
        {
          v16h bv;
          if (nvalid) {
            bv = *(const v16h*)(wRow + kbase + cb + c0);
            __builtin_prefetch(wRow + kbase + cb + c0 + 32, 0, 3);
          } else {
            #pragma unroll
            for (int j = 0; j < 16; ++j) bv[j] = (_Float16)0.0f;
          }
          *(v16h*)(&Bs[r * LDSF + c0]) = bv;
        }
        __syncthreads();

        // ---- fragments (b128 LDS loads) ----
        // A: lanes 0-15 hold K0-15; lanes 16-31 hold K16-31
        int ra0 = (warpM * 32 + lrow) * LDSF;
        int ra1 = (warpM * 32 + 16 + lrow) * LDSF;
        v8h a0lo = *(const v8h*)(&As[ra0 + sel * 16]);
        v8h a0hi = *(const v8h*)(&As[ra0 + sel * 16 + 8]);
        v8h a1lo = *(const v8h*)(&As[ra1 + sel * 16]);
        v8h a1hi = *(const v8h*)(&As[ra1 + sel * 16 + 8]);
        v16h a0v = __builtin_shufflevector(a0lo, a0hi, 0,1,2,3,4,5,6,7,8,9,10,11,12,13,14,15);
        v16h a1v = __builtin_shufflevector(a1lo, a1hi, 0,1,2,3,4,5,6,7,8,9,10,11,12,13,14,15);
        // B: lanes 0-15 hold K0-15 of column N=lane; lanes 16-31 hold K16-31
        v16h b0v = *(const v16h*)(&Bs[(warpN * 32 + lrow) * LDSF + sel * 16]);
        v16h b1v = *(const v16h*)(&Bs[(warpN * 32 + 16 + lrow) * LDSF + sel * 16]);

        acc[0][0] = __builtin_amdgcn_wmma_f32_16x16x32_f16(false, a0v, false, b0v, (short)0, acc[0][0], false, false);
        acc[0][1] = __builtin_amdgcn_wmma_f32_16x16x32_f16(false, a0v, false, b1v, (short)0, acc[0][1], false, false);
        acc[1][0] = __builtin_amdgcn_wmma_f32_16x16x32_f16(false, a1v, false, b0v, (short)0, acc[1][0], false, false);
        acc[1][1] = __builtin_amdgcn_wmma_f32_16x16x32_f16(false, a1v, false, b1v, (short)0, acc[1][1], false, false);
        __syncthreads();
      }
    }
  }
  conv_epilogue(p, acc, tileM, tileN, warpM, warpN, lane);
}

// ---------------- generic path (stem conv, Cin=3) ----------------
#define LDSS 34
__global__ __launch_bounds__(128)
void yolo_conv_wmma_gen(ConvParams p) {
  __shared__ _Float16 As[64][LDSS];
  __shared__ _Float16 Bs[64][LDSS];

  const int tid   = threadIdx.x;
  const int lane  = tid & 31;
  const int wave  = tid >> 5;
  const int warpM = wave >> 1;
  const int warpN = wave & 1;
  const int tileM = blockIdx.x * 64;
  const int tileN = blockIdx.y * 64;

  const int r  = tid >> 1;
  const int c0 = (tid & 1) * 16;

  const int m = tileM + r;
  const bool mvalid = (m < p.M);
  int ow = 0, oh = 0, bb = 0;
  if (mvalid) {
    ow = m % p.Wo;
    int t2 = m / p.Wo;
    oh = t2 % p.Ho;
    bb = t2 / p.Ho;
  }
  const int ihs = oh * p.stride - p.pad;
  const int iws = ow * p.stride - p.pad;
  const int n  = tileN + r;

  v8f zero = {0.f,0.f,0.f,0.f,0.f,0.f,0.f,0.f};
  v8f acc[2][2];
  acc[0][0] = zero; acc[0][1] = zero; acc[1][0] = zero; acc[1][1] = zero;

  for (int kk = 0; kk < p.Ktot; kk += 32) {
    #pragma unroll
    for (int c = 0; c < 16; ++c) {
      int k = kk + c0 + c;
      float v = 0.0f;
      if (mvalid && k < p.Ktot) {
        int ci = k % p.Cin;
        int kt = k / p.Cin;
        int kw = kt % p.Ksz;
        int kh = kt / p.Ksz;
        int ih = ihs + kh;
        int iw = iws + kw;
        if (ih >= 0 && ih < p.H && iw >= 0 && iw < p.W)
          v = p.x[((size_t)((bb * p.H + ih) * p.W + iw)) * p.Cin + ci];
      }
      As[r][c0 + c] = (_Float16)v;
    }
    #pragma unroll
    for (int c = 0; c < 16; ++c) {
      int k = kk + c0 + c;
      _Float16 v = (_Float16)0.0f;
      if (n < p.Cout && k < p.Ktot) v = p.wT[(size_t)n * p.Ktot + k];
      Bs[r][c0 + c] = v;
    }
    __syncthreads();

    const int lrow = lane & 15;
    const int sel  = lane >> 4;
    v16h a0v, a1v, b0v, b1v;
    #pragma unroll
    for (int j = 0; j < 8; ++j) {
      a0v[j]     = As[warpM * 32 + lrow][sel * 16 + j];
      a0v[8 + j] = As[warpM * 32 + lrow][sel * 16 + 8 + j];
      a1v[j]     = As[warpM * 32 + 16 + lrow][sel * 16 + j];
      a1v[8 + j] = As[warpM * 32 + 16 + lrow][sel * 16 + 8 + j];
    }
    #pragma unroll
    for (int j = 0; j < 16; ++j) {
      b0v[j] = Bs[warpN * 32 + lrow][sel * 16 + j];
      b1v[j] = Bs[warpN * 32 + 16 + lrow][sel * 16 + j];
    }

    acc[0][0] = __builtin_amdgcn_wmma_f32_16x16x32_f16(false, a0v, false, b0v, (short)0, acc[0][0], false, false);
    acc[0][1] = __builtin_amdgcn_wmma_f32_16x16x32_f16(false, a0v, false, b1v, (short)0, acc[0][1], false, false);
    acc[1][0] = __builtin_amdgcn_wmma_f32_16x16x32_f16(false, a1v, false, b0v, (short)0, acc[1][0], false, false);
    acc[1][1] = __builtin_amdgcn_wmma_f32_16x16x32_f16(false, a1v, false, b1v, (short)0, acc[1][1], false, false);
    __syncthreads();
  }
  conv_epilogue(p, acc, tileM, tileN, warpM, warpN, lane);
}

// fp32 HWIO (k,k,Cin,Cout) -> f16 [Cout][k*k*Cin]
__global__ void yolo_cvt_weight(const float* w, _Float16* wT, int Ktot, int Cout) {
  int total = Ktot * Cout;
  for (int i = blockIdx.x * blockDim.x + threadIdx.x; i < total;
       i += gridDim.x * blockDim.x) {
    int co = i / Ktot;
    int k  = i - co * Ktot;
    wT[i] = (_Float16)w[(size_t)k * Cout + co];
  }
}

// nearest 2x upsample of u (B,h,w,C1), concat skip (B,2h,2w,C2) -> (B,2h,2w,C1+C2)
__global__ void yolo_upcat(const float* u, const float* skip, float* out,
                           int B, int h, int w, int C1, int C2) {
  int C = C1 + C2;
  size_t total = (size_t)B * 2 * h * 2 * w * C;
  for (size_t idx = (size_t)blockIdx.x * blockDim.x + threadIdx.x; idx < total;
       idx += (size_t)gridDim.x * blockDim.x) {
    int c = (int)(idx % C);
    size_t t = idx / C;
    int X = (int)(t % (2 * w)); t /= (2 * w);
    int Y = (int)(t % (2 * h));
    int b = (int)(t / (2 * h));
    float v;
    if (c < C1)
      v = u[(((size_t)b * h + (Y >> 1)) * w + (X >> 1)) * C1 + c];
    else
      v = skip[(((size_t)b * 2 * h + Y) * 2 * w + X) * C2 + (c - C1)];
    out[idx] = v;
  }
}

// yolo head decode: raw (B,gs,gs,3,7) -> boxes (B,Ntot,4), scores=obj*cls (B,Ntot,2)
__global__ void yolo_decode(const float* pred, float* boxes, float* scores,
                            int B, int gs, int off, int Ntot,
                            float aw0, float ah0, float aw1, float ah1,
                            float aw2, float ah2) {
  int total = B * gs * gs * 3;
  for (int idx = blockIdx.x * blockDim.x + threadIdx.x; idx < total;
       idx += gridDim.x * blockDim.x) {
    const float* pp = pred + (size_t)idx * 7;
    int a = idx % 3;
    int rr = idx / 3;
    int xw = rr % gs; rr /= gs;
    int yh = rr % gs;
    int b  = rr / gs;
    float inv = 1.0f / (float)gs;
    float bx = (1.0f / (1.0f + expf(-pp[0])) + (float)xw) * inv;
    float by = (1.0f / (1.0f + expf(-pp[1])) + (float)yh) * inv;
    float aw = (a == 0) ? aw0 : ((a == 1) ? aw1 : aw2);
    float ah = (a == 0) ? ah0 : ((a == 1) ? ah1 : ah2);
    float bw = expf(pp[2]) * aw;
    float bh = expf(pp[3]) * ah;
    float obj = 1.0f / (1.0f + expf(-pp[4]));
    float s0 = obj / (1.0f + expf(-pp[5]));
    float s1 = obj / (1.0f + expf(-pp[6]));
    int li = idx - b * (gs * gs * 3);
    size_t nn = (size_t)b * Ntot + off + li;
    boxes[nn * 4 + 0] = bx - bw * 0.5f;
    boxes[nn * 4 + 1] = by - bh * 0.5f;
    boxes[nn * 4 + 2] = bx + bw * 0.5f;
    boxes[nn * 4 + 3] = by + bh * 0.5f;
    scores[nn * 2 + 0] = s0;
    scores[nn * 2 + 1] = s1;
  }
}

// one block per (image, class): 100 serial steps of argmax + IoU suppression
#define NMS_T 256
__global__ __launch_bounds__(NMS_T)
void yolo_nms(const float* boxes, const float* scores, float* scW,
              float* outB, float* outS, int Ntot, int C) {
  int b = blockIdx.x / C;
  int c = blockIdx.x % C;
  const float* bx = boxes + (size_t)b * Ntot * 4;
  float* sc = scW + ((size_t)b * C + c) * Ntot;
  for (int i = threadIdx.x; i < Ntot; i += NMS_T) {
    float s = scores[((size_t)b * Ntot + i) * C + c];
    sc[i] = (s > 0.05f) ? s : -1.0f;
  }
  __syncthreads();
  __shared__ float sVal[NMS_T];
  __shared__ int   sIdx[NMS_T];
  float* ob = outB + (((size_t)b * C + c) * 100) * 4;
  float* os = outS + ((size_t)b * C + c) * 100;
  for (int it = 0; it < 100; ++it) {
    float best = -1e30f;
    int bi = 0x7fffffff;
    for (int i = threadIdx.x; i < Ntot; i += NMS_T) {
      float s = sc[i];
      if (s > best) { best = s; bi = i; }
    }
    sVal[threadIdx.x] = best; sIdx[threadIdx.x] = bi;
    __syncthreads();
    for (int off = NMS_T / 2; off > 0; off >>= 1) {
      if (threadIdx.x < off) {
        float v2 = sVal[threadIdx.x + off];
        int   i2 = sIdx[threadIdx.x + off];
        if (v2 > sVal[threadIdx.x] ||
            (v2 == sVal[threadIdx.x] && i2 < sIdx[threadIdx.x])) {
          sVal[threadIdx.x] = v2; sIdx[threadIdx.x] = i2;
        }
      }
      __syncthreads();
    }
    int imax = sIdx[0];
    float smax = sVal[0];
    float b0 = bx[(size_t)imax * 4 + 0], b1 = bx[(size_t)imax * 4 + 1];
    float b2 = bx[(size_t)imax * 4 + 2], b3 = bx[(size_t)imax * 4 + 3];
    float areaI = (b2 - b0) * (b3 - b1);
    for (int i = threadIdx.x; i < Ntot; i += NMS_T) {
      float x1 = bx[(size_t)i * 4 + 0], y1 = bx[(size_t)i * 4 + 1];
      float x2 = bx[(size_t)i * 4 + 2], y2 = bx[(size_t)i * 4 + 3];
      float ix1 = fmaxf(b0, x1), iy1 = fmaxf(b1, y1);
      float ix2 = fminf(b2, x2), iy2 = fminf(b3, y2);
      float inter = fmaxf(ix2 - ix1, 0.0f) * fmaxf(iy2 - iy1, 0.0f);
      float areas = (x2 - x1) * (y2 - y1);
      float iou = inter / (areaI + areas - inter + 1e-9f);
      float s = sc[i];
      if (iou > 0.5f) s = -1.0f;
      if (i == imax)  s = -1.0f;
      sc[i] = s;
    }
    if (threadIdx.x == 0) {
      bool valid = smax > 0.0f;
      ob[it * 4 + 0] = valid ? b0 : 0.0f;
      ob[it * 4 + 1] = valid ? b1 : 0.0f;
      ob[it * 4 + 2] = valid ? b2 : 0.0f;
      ob[it * 4 + 3] = valid ? b3 : 0.0f;
      os[it] = valid ? smax : -1.0f;
    }
    __syncthreads();
  }
}

// one block per image: stable top-100 of C*100 candidates, final output pack
__global__ void yolo_topk(const float* nmsB, const float* nmsS, float* out, int B, int C) {
  int b = blockIdx.x;
  int NC = C * 100;
  __shared__ float cs[256];
  for (int i = threadIdx.x; i < NC; i += blockDim.x)
    cs[i] = nmsS[(size_t)b * NC + i];
  __syncthreads();
  if (threadIdx.x == 0) {
    float* selB = out + (size_t)b * 400;
    float* topS = out + (size_t)B * 400 + (size_t)b * 100;
    float* selC = out + (size_t)B * 400 + (size_t)B * 100 + (size_t)b * 100;
    float* cntO = out + (size_t)B * 400 + (size_t)B * 200 + b;
    float cnt = 0.0f;
    for (int it = 0; it < 100; ++it) {
      float best = cs[0]; int bi = 0;
      for (int i = 1; i < NC; ++i)
        if (cs[i] > best) { best = cs[i]; bi = i; }
      cs[bi] = -1e30f;
      float valid = (best > 0.0f) ? 1.0f : 0.0f;
      const float* bbp = nmsB + ((size_t)b * NC + bi) * 4;
      for (int k = 0; k < 4; ++k)
        selB[it * 4 + k] = fminf(fmaxf(bbp[k], 0.0f), 1.0f) * valid;
      topS[it] = fmaxf(best, 0.0f);
      selC[it] = (float)(bi / 100) * valid;
      cnt += valid;
    }
    *cntO = cnt;
  }
}

// ---------------------------------------------------------------------------
// Host orchestration
// ---------------------------------------------------------------------------
namespace {

struct Ctx {
  hipStream_t stream;
  char*  base;
  size_t off;
};

static void* walloc(Ctx& c, size_t bytes) {
  c.off = (c.off + 255) & ~(size_t)255;
  void* p = c.base + c.off;
  c.off += bytes;
  return p;
}

struct PW { void* const* din; int ix; };
static inline const float* nx(PW& p) { return (const float*)p.din[p.ix++]; }

static void runConv(Ctx& ctx, const float* x, const float* w, const float* s,
                    const float* b, const float* residual, float* y,
                    int B, int H, int W, int Cin, int Cout, int Ksz, int stride) {
  ConvParams p;
  p.x = x; p.scale = s; p.bias = b; p.residual = residual; p.y = y;
  p.B = B; p.H = H; p.W = W; p.Cin = Cin; p.Cout = Cout;
  p.Ksz = Ksz; p.stride = stride;
  p.pad = (Ksz == 3) ? 1 : 0;
  p.Ho = (stride == 2) ? H / 2 : H;
  p.Wo = (stride == 2) ? W / 2 : W;
  p.Ktot = Ksz * Ksz * Cin;
  p.M = B * p.Ho * p.Wo;
  p.bnAct = (s != nullptr) ? 1 : 0;
  _Float16* wT = (_Float16*)walloc(ctx, (size_t)p.Ktot * Cout * sizeof(_Float16));
  p.wT = wT;
  int tw = p.Ktot * Cout;
  int cg = (tw + 255) / 256; if (cg > 4096) cg = 4096;
  yolo_cvt_weight<<<cg, 256, 0, ctx.stream>>>(w, wT, p.Ktot, Cout);
  dim3 g((p.M + 63) / 64, (Cout + 63) / 64);
  if ((Cin & 31) == 0)
    yolo_conv_wmma_fast<<<g, 128, 0, ctx.stream>>>(p);
  else
    yolo_conv_wmma_gen<<<g, 128, 0, ctx.stream>>>(p);
}

} // namespace

extern "C" void kernel_launch(void* const* d_in, const int* in_sizes, int n_in,
                              void* d_out, int out_size, void* d_ws, size_t ws_size,
                              hipStream_t stream) {
  (void)n_in; (void)out_size; (void)ws_size;
  const int B = in_sizes[0] / (256 * 256 * 3);
  const int Ntot = 3 * (32 * 32 + 64 * 64 + 128 * 128); // 64512
  const int C = 2;

  Ctx ctx; ctx.stream = stream; ctx.base = (char*)d_ws; ctx.off = 0;

  // activation / detection buffers
  size_t nS = (size_t)B * 128 * 128 * 256;
  float* S1   = (float*)walloc(ctx, nS * sizeof(float));
  float* S2   = (float*)walloc(ctx, nS * sizeof(float));
  float* S3   = (float*)walloc(ctx, nS * sizeof(float));
  float* x36  = (float*)walloc(ctx, (size_t)B * 128 * 128 * 128 * sizeof(float));
  float* x61  = (float*)walloc(ctx, (size_t)B * 64 * 64 * 128 * sizeof(float));
  float* o1r  = (float*)walloc(ctx, (size_t)B * 32 * 32 * 21 * sizeof(float));
  float* o2r  = (float*)walloc(ctx, (size_t)B * 64 * 64 * 21 * sizeof(float));
  float* o3r  = (float*)walloc(ctx, (size_t)B * 128 * 128 * 21 * sizeof(float));
  float* boxes  = (float*)walloc(ctx, (size_t)B * Ntot * 4 * sizeof(float));
  float* scores = (float*)walloc(ctx, (size_t)B * Ntot * 2 * sizeof(float));
  float* scW    = (float*)walloc(ctx, (size_t)B * C * Ntot * sizeof(float));
  float* nmsB   = (float*)walloc(ctx, (size_t)B * C * 100 * 4 * sizeof(float));
  float* nmsS   = (float*)walloc(ctx, (size_t)B * C * 100 * sizeof(float));

  PW pw{d_in, 0};
  const float* xin = nx(pw);

  // dnconv1: 3x3 s1, 3->32 @256 (generic path)
  { const float *w = nx(pw), *s = nx(pw), *b = nx(pw);
    runConv(ctx, xin, w, s, b, nullptr, S1, B, 256, 256, 3, 32, 3, 1); }

  // b3: down 32->128 s2 @256->128, 8 res (128->64 1x1, 64->128 3x3 +res)
  { const float *w = nx(pw), *s = nx(pw), *b = nx(pw);
    runConv(ctx, S1, w, s, b, nullptr, S2, B, 256, 256, 32, 128, 3, 2); }
  {
    float* t = S2; float* other = S1;
    for (int i = 0; i < 8; ++i) {
      const float *w1 = nx(pw), *s1 = nx(pw), *b1 = nx(pw);
      runConv(ctx, t, w1, s1, b1, nullptr, S3, B, 128, 128, 128, 64, 1, 1);
      float* outb = (i == 7) ? x36 : other;
      const float *w2 = nx(pw), *s2 = nx(pw), *b2 = nx(pw);
      runConv(ctx, S3, w2, s2, b2, t, outb, B, 128, 128, 64, 128, 3, 1);
      other = t; t = outb;
    }
  }

  // b4: down 128->128 s2 @128->64, 8 res
  { const float *w = nx(pw), *s = nx(pw), *b = nx(pw);
    runConv(ctx, x36, w, s, b, nullptr, S1, B, 128, 128, 128, 128, 3, 2); }
  {
    float* t = S1; float* other = S2;
    for (int i = 0; i < 8; ++i) {
      const float *w1 = nx(pw), *s1 = nx(pw), *b1 = nx(pw);
      runConv(ctx, t, w1, s1, b1, nullptr, S3, B, 64, 64, 128, 64, 1, 1);
      float* outb = (i == 7) ? x61 : other;
      const float *w2 = nx(pw), *s2 = nx(pw), *b2 = nx(pw);
      runConv(ctx, S3, w2, s2, b2, t, outb, B, 64, 64, 64, 128, 3, 1);
      other = t; t = outb;
    }
  }

  // b5: down 128->256 s2 @64->32, 4 res (256->128, 128->256)
  { const float *w = nx(pw), *s = nx(pw), *b = nx(pw);
    runConv(ctx, x61, w, s, b, nullptr, S1, B, 64, 64, 128, 256, 3, 2); }
  float* t5 = S1;
  {
    float* t = S1; float* other = S2;
    for (int i = 0; i < 4; ++i) {
      const float *w1 = nx(pw), *s1 = nx(pw), *b1 = nx(pw);
      runConv(ctx, t, w1, s1, b1, nullptr, S3, B, 32, 32, 256, 128, 1, 1);
      float* outb = other;
      const float *w2 = nx(pw), *s2 = nx(pw), *b2 = nx(pw);
      runConv(ctx, S3, w2, s2, b2, t, outb, B, 32, 32, 128, 256, 3, 1);
      other = t; t = outb;
    }
    t5 = t; // ends at S1
  }

  // yc1 @32: 256->512, 512->1024, 1024->512, 512->1024, 1024->512
  { const float *w = nx(pw), *s = nx(pw), *b = nx(pw);
    runConv(ctx, t5, w, s, b, nullptr, S2, B, 32, 32, 256, 512, 1, 1); }
  { const float *w = nx(pw), *s = nx(pw), *b = nx(pw);
    runConv(ctx, S2, w, s, b, nullptr, S3, B, 32, 32, 512, 1024, 3, 1); }
  { const float *w = nx(pw), *s = nx(pw), *b = nx(pw);
    runConv(ctx, S3, w, s, b, nullptr, S2, B, 32, 32, 1024, 512, 1, 1); }
  { const float *w = nx(pw), *s = nx(pw), *b = nx(pw);
    runConv(ctx, S2, w, s, b, nullptr, S3, B, 32, 32, 512, 1024, 3, 1); }
  { const float *w = nx(pw), *s = nx(pw), *b = nx(pw);
    runConv(ctx, S3, w, s, b, nullptr, S2, B, 32, 32, 1024, 512, 1, 1); }
  float* y1 = S2;

  // out1: 3x3 512->1024 bn, 1x1 1024->21 bias
  { const float *w = nx(pw), *s = nx(pw), *b = nx(pw);
    runConv(ctx, y1, w, s, b, nullptr, S3, B, 32, 32, 512, 1024, 3, 1); }
  { const float *w = nx(pw), *b = nx(pw);
    runConv(ctx, S3, w, nullptr, b, nullptr, o1r, B, 32, 32, 1024, 21, 1, 1); }

  // yc2: pre 1x1 512->256, up+cat(x61) -> 384 @64, then 5 convs
  { const float *w = nx(pw), *s = nx(pw), *b = nx(pw);
    runConv(ctx, y1, w, s, b, nullptr, S1, B, 32, 32, 512, 256, 1, 1); }
  { size_t tot = (size_t)B * 64 * 64 * 384;
    int g = (int)((tot + 255) / 256); if (g > 4096) g = 4096;
    yolo_upcat<<<g, 256, 0, stream>>>(S1, x61, S3, B, 32, 32, 256, 128); }
  { const float *w = nx(pw), *s = nx(pw), *b = nx(pw);
    runConv(ctx, S3, w, s, b, nullptr, S1, B, 64, 64, 384, 256, 1, 1); }
  { const float *w = nx(pw), *s = nx(pw), *b = nx(pw);
    runConv(ctx, S1, w, s, b, nullptr, S3, B, 64, 64, 256, 512, 3, 1); }
  { const float *w = nx(pw), *s = nx(pw), *b = nx(pw);
    runConv(ctx, S3, w, s, b, nullptr, S1, B, 64, 64, 512, 256, 1, 1); }
  { const float *w = nx(pw), *s = nx(pw), *b = nx(pw);
    runConv(ctx, S1, w, s, b, nullptr, S3, B, 64, 64, 256, 512, 3, 1); }
  { const float *w = nx(pw), *s = nx(pw), *b = nx(pw);
    runConv(ctx, S3, w, s, b, nullptr, S1, B, 64, 64, 512, 256, 1, 1); }
  float* y2 = S1;

  // out2
  { const float *w = nx(pw), *s = nx(pw), *b = nx(pw);
    runConv(ctx, y2, w, s, b, nullptr, S3, B, 64, 64, 256, 512, 3, 1); }
  { const float *w = nx(pw), *b = nx(pw);
    runConv(ctx, S3, w, nullptr, b, nullptr, o2r, B, 64, 64, 512, 21, 1, 1); }

  // yc3: pre 1x1 256->128, up+cat(x36) -> 256 @128, then 5 convs
  { const float *w = nx(pw), *s = nx(pw), *b = nx(pw);
    runConv(ctx, y2, w, s, b, nullptr, S2, B, 64, 64, 256, 128, 1, 1); }
  { size_t tot = (size_t)B * 128 * 128 * 256;
    int g = (int)((tot + 255) / 256); if (g > 4096) g = 4096;
    yolo_upcat<<<g, 256, 0, stream>>>(S2, x36, S3, B, 64, 64, 128, 128); }
  { const float *w = nx(pw), *s = nx(pw), *b = nx(pw);
    runConv(ctx, S3, w, s, b, nullptr, S1, B, 128, 128, 256, 128, 1, 1); }
  { const float *w = nx(pw), *s = nx(pw), *b = nx(pw);
    runConv(ctx, S1, w, s, b, nullptr, S3, B, 128, 128, 128, 256, 3, 1); }
  { const float *w = nx(pw), *s = nx(pw), *b = nx(pw);
    runConv(ctx, S3, w, s, b, nullptr, S1, B, 128, 128, 256, 128, 1, 1); }
  { const float *w = nx(pw), *s = nx(pw), *b = nx(pw);
    runConv(ctx, S1, w, s, b, nullptr, S3, B, 128, 128, 128, 256, 3, 1); }
  { const float *w = nx(pw), *s = nx(pw), *b = nx(pw);
    runConv(ctx, S3, w, s, b, nullptr, S1, B, 128, 128, 256, 128, 1, 1); }
  float* y3 = S1;

  // out3
  { const float *w = nx(pw), *s = nx(pw), *b = nx(pw);
    runConv(ctx, y3, w, s, b, nullptr, S3, B, 128, 128, 128, 256, 3, 1); }
  { const float *w = nx(pw), *b = nx(pw);
    runConv(ctx, S3, w, nullptr, b, nullptr, o3r, B, 128, 128, 256, 21, 1, 1); }

  // decode: anchors /416; o1 uses 6,7,8; o2: 3,4,5; o3: 0,1,2
  const float AW[9] = {10, 16, 33, 30, 62, 59, 116, 156, 373};
  const float AH[9] = {13, 30, 23, 61, 45, 119, 90, 198, 326};
  {
    int tot = B * 32 * 32 * 3, g = (tot + 255) / 256;
    yolo_decode<<<g, 256, 0, stream>>>(o1r, boxes, scores, B, 32, 0, Ntot,
        AW[6]/416.f, AH[6]/416.f, AW[7]/416.f, AH[7]/416.f, AW[8]/416.f, AH[8]/416.f);
  }
  {
    int tot = B * 64 * 64 * 3, g = (tot + 255) / 256;
    yolo_decode<<<g, 256, 0, stream>>>(o2r, boxes, scores, B, 64, 3072, Ntot,
        AW[3]/416.f, AH[3]/416.f, AW[4]/416.f, AH[4]/416.f, AW[5]/416.f, AH[5]/416.f);
  }
  {
    int tot = B * 128 * 128 * 3, g = (tot + 255) / 256;
    yolo_decode<<<g, 256, 0, stream>>>(o3r, boxes, scores, B, 128, 15360, Ntot,
        AW[0]/416.f, AH[0]/416.f, AW[1]/416.f, AH[1]/416.f, AW[2]/416.f, AH[2]/416.f);
  }

  yolo_nms<<<B * C, NMS_T, 0, stream>>>(boxes, scores, scW, nmsB, nmsS, Ntot, C);
  yolo_topk<<<B, 256, 0, stream>>>(nmsB, nmsS, (float*)d_out, B, C);
}
